// MultiHeadSelfAttention_50294067036437
// MI455X (gfx1250) — matrix operand order repaired and checked by reading, not compiled, not run on hardware
//
#include <hip/hip_runtime.h>

// ---------------------------------------------------------------------------
// Types for CDNA5 WMMA (gfx1250, wave32)
// ---------------------------------------------------------------------------
typedef __attribute__((ext_vector_type(16))) __bf16 v16bf;
typedef __attribute__((ext_vector_type(8)))  float  v8f;

union FragB16 {
    v16bf v;
    uint4 q[2];
    unsigned short u[16];
};

__device__ __forceinline__ unsigned short f2bf(float f) {
    union { float f; unsigned u; } c;
    c.f = f;
    unsigned u = c.u;
    unsigned r = (u + 0x7FFFu + ((u >> 16) & 1u)) >> 16;  // RNE
    return (unsigned short)r;
}

__device__ __forceinline__ v8f wmma_bf16(v16bf a, v16bf b, v8f c) {
    return __builtin_amdgcn_wmma_f32_16x16x32_bf16(false, a, false, b,
                                                   (short)0, c, false, false);
}

__device__ __forceinline__ v8f v8f_zero() {
    v8f z = {0.f, 0.f, 0.f, 0.f, 0.f, 0.f, 0.f, 0.f};
    return z;
}

// A fragment: p points at (row, k0 + half*8); elements 8..15 are at +16
__device__ __forceinline__ void loadFragA(FragB16& f, const unsigned short* p) {
    f.q[0] = *(const uint4*)(p);
    f.q[1] = *(const uint4*)(p + 16);
}
// B fragment: p points at (col, k0 + half*8); elements 8..15 are at +16
__device__ __forceinline__ void loadFragB(FragB16& f, const unsigned short* p) {
    f.q[0] = *(const uint4*)(p);
    f.q[1] = *(const uint4*)(p + 16);
}

// ---------------------------------------------------------------------------
// Problem constants
// ---------------------------------------------------------------------------
static constexpr int Bsz = 4;
static constexpr int S   = 2048;
static constexpr int Dm  = 1024;
static constexpr int H   = 16;
static constexpr int DH  = 64;
static constexpr int M_ROWS = Bsz * S;     // 8192

// ---------------------------------------------------------------------------
// fp32 -> bf16 conversion kernels
// ---------------------------------------------------------------------------
__global__ void cvt_f32_bf16(const float* __restrict__ src,
                             unsigned short* __restrict__ dst, int n) {
    int i = blockIdx.x * blockDim.x + threadIdx.x;
    if (i < n) dst[i] = f2bf(src[i]);
}

// w: [K=1024][N=1024] row-major fp32  ->  wT: [N][K] bf16
__global__ void cvt_transpose_w(const float* __restrict__ w,
                                unsigned short* __restrict__ wT) {
    int i = blockIdx.x * blockDim.x + threadIdx.x; // 0 .. 1048575
    int k = i >> 10;
    int n = i & 1023;
    wT[(size_t)n * Dm + k] = f2bf(w[i]);
}

// ---------------------------------------------------------------------------
// GEMM: C[row, col] = A[row, :] . BT[col, :] + bias[col]
//   A  : bf16 [M_ROWS, 1024] row-major
//   BT : bf16 [1024, 1024]  (weight pre-transposed: BT[n][k])
//   mode 0: write bf16 to [B,H,S,dh]   (Q / K)
//   mode 1: write bf16 to [B,H,dh,S]   (V transposed)
//   mode 2: write fp32 to [B,S,D]      (final output)
// Block: 256 threads = 8 waves; wave -> 32 rows x 64 cols; block -> 256 rows.
// Software-pipelined: fragments for k-step i+1 in flight during WMMAs of i.
// ---------------------------------------------------------------------------
__global__ void __launch_bounds__(256)
gemm_bf16(const unsigned short* __restrict__ A,
          const unsigned short* __restrict__ BT,
          const float* __restrict__ bias,
          void* __restrict__ outp, int mode) {
    const int tid  = threadIdx.x;
    const int wave = tid >> 5;
    const int lane = tid & 31;
    const int half = lane >> 4;
    const int nl   = lane & 15;

    const int rowBase = blockIdx.x * 256 + wave * 32;
    const int colBase = blockIdx.y * 64;

    v8f acc[2][4];
#pragma unroll
    for (int mi = 0; mi < 2; ++mi)
#pragma unroll
        for (int t = 0; t < 4; ++t) acc[mi][t] = v8f_zero();

    // per-lane fragment base pointers (k-offset folded in per step)
    const unsigned short* aP[2];
    aP[0] = A + (size_t)(rowBase + nl) * Dm + half * 8;
    aP[1] = A + (size_t)(rowBase + 16 + nl) * Dm + half * 8;
    const unsigned short* bP[4];
#pragma unroll
    for (int t = 0; t < 4; ++t)
        bP[t] = BT + (size_t)(colBase + t * 16 + nl) * Dm + half * 8;

    FragB16 aC[2], bC[4], aN[2], bN[4];
    loadFragA(aC[0], aP[0]);
    loadFragA(aC[1], aP[1]);
#pragma unroll
    for (int t = 0; t < 4; ++t) loadFragB(bC[t], bP[t]);

    for (int kb = 0; kb < Dm; kb += 64) {
        // prefetch k-step kb+32 while computing kb
        loadFragA(aN[0], aP[0] + kb + 32);
        loadFragA(aN[1], aP[1] + kb + 32);
#pragma unroll
        for (int t = 0; t < 4; ++t) loadFragB(bN[t], bP[t] + kb + 32);
#pragma unroll
        for (int t = 0; t < 4; ++t) {
            acc[0][t] = wmma_bf16(aC[0].v, bC[t].v, acc[0][t]);
            acc[1][t] = wmma_bf16(aC[1].v, bC[t].v, acc[1][t]);
        }
        // prefetch k-step kb+64 while computing kb+32
        if (kb + 64 < Dm) {
            loadFragA(aC[0], aP[0] + kb + 64);
            loadFragA(aC[1], aP[1] + kb + 64);
#pragma unroll
            for (int t = 0; t < 4; ++t) loadFragB(bC[t], bP[t] + kb + 64);
        }
#pragma unroll
        for (int t = 0; t < 4; ++t) {
            acc[0][t] = wmma_bf16(aN[0].v, bN[t].v, acc[0][t]);
            acc[1][t] = wmma_bf16(aN[1].v, bN[t].v, acc[1][t]);
        }
    }

    // Epilogue: C layout -> VGPR r holds row m = half*8 + r, col = nl
#pragma unroll
    for (int mi = 0; mi < 2; ++mi) {
#pragma unroll
        for (int r = 0; r < 8; ++r) {
            const int row = rowBase + mi * 16 + half * 8 + r;
            const int b   = row >> 11;        // / S
            const int s   = row & (S - 1);
#pragma unroll
            for (int t = 0; t < 4; ++t) {
                const int col = colBase + t * 16 + nl;
                float val = acc[mi][t][r] + bias[col];
                if (mode == 2) {
                    ((float*)outp)[(size_t)row * Dm + col] = val;
                } else {
                    const int h = col >> 6;
                    const int d = col & 63;
                    unsigned short* o16 = (unsigned short*)outp;
                    if (mode == 0) {
                        // [B,H,S,dh]
                        o16[((size_t)(b * H + h) * S + s) * DH + d] = f2bf(val);
                    } else {
                        // [B,H,dh,S]  (V transposed for PV B-fragments)
                        o16[((size_t)(b * H + h) * DH + d) * S + s] = f2bf(val);
                    }
                }
            }
        }
    }
}

// ---------------------------------------------------------------------------
// Flash attention: per (b,h), online-softmax over keys.
//   Q,K : bf16 [B,H,S,dh]
//   VT  : bf16 [B,H,dh,S]
//   O   : bf16 [B,S,D]   (ready as A-matrix for the output projection)
// Block: 256 threads = 8 waves; wave -> 16 queries; block -> 128 queries.
// Grid: (B*H, S/128)
// Pipelining: V fragments for chunk kc are issued at iteration top (latency
// hidden by softmax); K fragments for kc+32 are prefetched right after the
// score WMMAs (latency hidden by softmax + LDS transpose).
// ---------------------------------------------------------------------------
__global__ void __launch_bounds__(256)
attn_kernel(const unsigned short* __restrict__ Q,
            const unsigned short* __restrict__ K,
            const unsigned short* __restrict__ VT,
            unsigned short* __restrict__ O) {
    __shared__ __align__(16) unsigned short ldsP[8][16][32]; // 8KB: P transpose

    const int tid  = threadIdx.x;
    const int wave = tid >> 5;
    const int lane = tid & 31;
    const int half = lane >> 4;
    const int nl   = lane & 15;

    const int bh    = blockIdx.x;                  // b*H + h
    const int qBase = blockIdx.y * 128 + wave * 16;
    const size_t headOff = (size_t)bh * S * DH;

    // Q fragments (A layout), dh=64 -> 2 k-steps of 32
    FragB16 qA[2];
    {
        const unsigned short* qRow =
            Q + headOff + (size_t)(qBase + nl) * DH + half * 8;
        loadFragA(qA[0], qRow);
        loadFragA(qA[1], qRow + 32);
    }

    // per-lane fragment base pointers
    const unsigned short* kP0 = K + headOff + (size_t)nl * DH + half * 8;
    const unsigned short* kP1 = K + headOff + (size_t)(16 + nl) * DH + half * 8;
    const unsigned short* vP[4];
#pragma unroll
    for (int t = 0; t < 4; ++t)
        vP[t] = VT + headOff + (size_t)(t * 16 + nl) * S + half * 8;

    v8f oAcc[4];
#pragma unroll
    for (int t = 0; t < 4; ++t) oAcc[t] = v8f_zero();
    float mRun[8], lRun[8];
#pragma unroll
    for (int r = 0; r < 8; ++r) { mRun[r] = -3.0e38f; lRun[r] = 0.f; }

    const float scale = 0.125f; // 1/sqrt(64)

    // preload K fragments for the first chunk: [kTile(2)][ks(2)]
    FragB16 kB[2][2];
#pragma unroll
    for (int ks = 0; ks < 2; ++ks) {
        loadFragB(kB[0][ks], kP0 + ks * 32);
        loadFragB(kB[1][ks], kP1 + ks * 32);
    }

    for (int kc = 0; kc < S; kc += 32) {
        // issue V fragment loads early; consumed only after the softmax
        FragB16 vB[4];
#pragma unroll
        for (int t = 0; t < 4; ++t) loadFragB(vB[t], vP[t] + kc);

        // ---- scores: S[16 q x 32 k] --------------------------------------
        v8f s0 = v8f_zero();
        v8f s1 = v8f_zero();
#pragma unroll
        for (int ks = 0; ks < 2; ++ks) {
            s0 = wmma_bf16(qA[ks].v, kB[0][ks].v, s0);
            s1 = wmma_bf16(qA[ks].v, kB[1][ks].v, s1);
        }

        // prefetch K fragments for the next chunk (latency hidden below)
        if (kc + 32 < S) {
            const size_t nk = (size_t)(kc + 32) * DH;
#pragma unroll
            for (int ks = 0; ks < 2; ++ks) {
                loadFragB(kB[0][ks], kP0 + nk + ks * 32);
                loadFragB(kB[1][ks], kP1 + nk + ks * 32);
            }
        }

        // ---- online softmax update (fp32, per-row across 16-lane half) ----
#pragma unroll
        for (int r = 0; r < 8; ++r) {
            float a = s0[r] * scale;
            float b = s1[r] * scale;
            float rm = fmaxf(a, b);
            rm = fmaxf(rm, __shfl_xor(rm, 1, 32));
            rm = fmaxf(rm, __shfl_xor(rm, 2, 32));
            rm = fmaxf(rm, __shfl_xor(rm, 4, 32));
            rm = fmaxf(rm, __shfl_xor(rm, 8, 32));
            const float mNew = fmaxf(mRun[r], rm);
            const float p0 = __expf(a - mNew);
            const float p1 = __expf(b - mNew);
            float rs = p0 + p1;
            rs += __shfl_xor(rs, 1, 32);
            rs += __shfl_xor(rs, 2, 32);
            rs += __shfl_xor(rs, 4, 32);
            rs += __shfl_xor(rs, 8, 32);
            const float resc = __expf(mRun[r] - mNew);
            lRun[r] = lRun[r] * resc + rs;
            mRun[r] = mNew;
#pragma unroll
            for (int t = 0; t < 4; ++t) oAcc[t][r] *= resc;
            // stage P (bf16) for the C->A transpose, row m = half*8 + r
            ldsP[wave][half * 8 + r][nl]      = f2bf(p0);
            ldsP[wave][half * 8 + r][16 + nl] = f2bf(p1);
        }
        __syncthreads();

        // ---- reload P as an A fragment (16x32) ----------------------------
        FragB16 pA;
        pA.q[0] = *(const uint4*)(&ldsP[wave][nl][half * 8]);
        pA.q[1] = *(const uint4*)(&ldsP[wave][nl][16 + half * 8]);
        __syncthreads();

        // ---- O += P . V  (V fragments were issued at iteration top) -------
#pragma unroll
        for (int t = 0; t < 4; ++t)
            oAcc[t] = wmma_bf16(pA.v, vB[t].v, oAcc[t]);
    }

    // ---- epilogue: O / l, write bf16 [B,S,D] ------------------------------
    const int b = bh >> 4;
    const int h = bh & 15;
#pragma unroll
    for (int r = 0; r < 8; ++r) {
        const float invl = 1.0f / lRun[r];
        const int row = b * S + qBase + half * 8 + r;
#pragma unroll
        for (int t = 0; t < 4; ++t) {
            O[(size_t)row * Dm + h * 64 + t * 16 + nl] =
                f2bf(oAcc[t][r] * invl);
        }
    }
}

// ---------------------------------------------------------------------------
// Launch
// ---------------------------------------------------------------------------
extern "C" void kernel_launch(void* const* d_in, const int* in_sizes, int n_in,
                              void* d_out, int out_size, void* d_ws, size_t ws_size,
                              hipStream_t stream) {
    const float* x  = (const float*)d_in[0];
    const float* wq = (const float*)d_in[1];
    const float* bq = (const float*)d_in[2];
    const float* wk = (const float*)d_in[3];
    const float* bk = (const float*)d_in[4];
    const float* wv = (const float*)d_in[5];
    const float* bv = (const float*)d_in[6];
    const float* wo = (const float*)d_in[7];
    const float* bo = (const float*)d_in[8];
    float* out = (float*)d_out;

    // workspace layout (ushort elements)
    unsigned short* ws = (unsigned short*)d_ws;
    const size_t nXD = (size_t)M_ROWS * Dm;   // 8M elems
    const size_t nW  = (size_t)Dm * Dm;       // 1M elems
    unsigned short* xb  = ws;                 size_t off = nXD;
    unsigned short* wqT = ws + off;           off += nW;
    unsigned short* wkT = ws + off;           off += nW;
    unsigned short* wvT = ws + off;           off += nW;
    unsigned short* woT = ws + off;           off += nW;
    unsigned short* qw  = ws + off;           off += nXD;
    unsigned short* kw  = ws + off;           off += nXD;
    unsigned short* vtw = ws + off;           off += nXD;
    unsigned short* ow  = ws + off;           off += nXD;

    // 1) conversions
    {
        int n = (int)nXD;
        cvt_f32_bf16<<<(n + 255) / 256, 256, 0, stream>>>(x, xb, n);
        cvt_transpose_w<<<(int)(nW / 256), 256, 0, stream>>>(wq, wqT);
        cvt_transpose_w<<<(int)(nW / 256), 256, 0, stream>>>(wk, wkT);
        cvt_transpose_w<<<(int)(nW / 256), 256, 0, stream>>>(wv, wvT);
        cvt_transpose_w<<<(int)(nW / 256), 256, 0, stream>>>(wo, woT);
    }

    // 2) Q/K/V projections (WMMA bf16)
    dim3 gGemm(M_ROWS / 256, Dm / 64);
    gemm_bf16<<<gGemm, 256, 0, stream>>>(xb, wqT, bq, qw, 0);
    gemm_bf16<<<gGemm, 256, 0, stream>>>(xb, wkT, bk, kw, 0);
    gemm_bf16<<<gGemm, 256, 0, stream>>>(xb, wvT, bv, vtw, 1);

    // 3) flash attention
    dim3 gAttn(Bsz * H, S / 128);
    attn_kernel<<<gAttn, 256, 0, stream>>>(qw, kw, vtw, ow);

    // 4) output projection -> fp32 d_out
    gemm_bf16<<<gGemm, 256, 0, stream>>>(ow, woT, bo, (void*)out, 2);
}
